// HeadLoraLinear_48421461295878
// MI455X (gfx1250) — matrix operand order repaired and checked by reading, not compiled, not run on hardware
//
#include <hip/hip_runtime.h>

// ---------------------------------------------------------------------------
// LoRA linear:  out = x @ W^T + 2.0 * (x @ A^T) @ B^T
//   x      [2048, 2048]  f32
//   W      [65536, 2048] f32   (row-major [out, in])
//   lora_A [64, 2048]    f32
//   lora_B [65536, 64]   f32
//   out    [2048, 65536] f32
// Strategy: K-extension fusion (K' = 2048 + 64) + bf16x3 split WMMA.
//   - branch-free main loop (63 iters), peeled tail (base#63, lora0, lora1)
//   - ext_vector float4 staging -> guaranteed global_load_b128
//   - v_cvt_pk_bf16_f32 hi/lo split (manual RNE fallback)
//   - m-fastest raster for W-tile L2 reuse; non-temporal output stores
// ---------------------------------------------------------------------------

#define M_TOK   2048
#define IN_F    2048
#define OUT_F   65536
#define RANK    64
#define SCALING 2.0f      // alpha/rank = 128/64

#define BM 128
#define BN 128
#define BK 32
#define PITCH 18                 // uints per LDS row: 16 data (32 bf16) + 2 pad
#define A_PART (BM * PITCH)      // 2304 uints per hi/lo part
#define NBASE (IN_F / BK)        // 64 base stages
#define NLORA (RANK / BK)        // 2 lora stages

typedef __attribute__((ext_vector_type(16))) __bf16        v16bf;
typedef __attribute__((ext_vector_type(2)))  __bf16        v2bf;
typedef __attribute__((ext_vector_type(8)))  float         v8f;
typedef __attribute__((ext_vector_type(4)))  float         f4v;
typedef __attribute__((ext_vector_type(4)))  unsigned int  uint4v;

union Frag {
    struct { uint4v q0, q1; };   // q0 -> VGPRs 0..3, q1 -> VGPRs 4..7
    v16bf v;
};

__device__ __forceinline__ unsigned short f2bf(float f) {
    unsigned int u = __float_as_uint(f);
    u += 0x7FFFu + ((u >> 16) & 1u);        // round to nearest even
    return (unsigned short)(u >> 16);
}

// pack two floats into one uint of 2 bf16 (RNE)
__device__ __forceinline__ unsigned int pk_bf16(float a, float b) {
#if __has_builtin(__builtin_amdgcn_cvt_pk_bf16_f32)
    v2bf h = __builtin_amdgcn_cvt_pk_bf16_f32(a, b);
    return __builtin_bit_cast(unsigned int, h);
#else
    return (unsigned int)f2bf(a) | ((unsigned int)f2bf(b) << 16);
#endif
}

// split a pair of floats into packed hi-bf16 word and packed lo-bf16 word
__device__ __forceinline__ void splitPair(float a, float b,
                                          unsigned int& hiw, unsigned int& low) {
    hiw = pk_bf16(a, b);
    float ha = __uint_as_float(hiw << 16);          // bf16(a) as f32
    float hb = __uint_as_float(hiw & 0xFFFF0000u);  // bf16(b) as f32
    low = pk_bf16(a - ha, b - hb);                  // residuals
}

// ---------------------------------------------------------------------------
// Kernel 1: low[m, r] = SCALING * dot(x[m, :], lora_A[r, :])     [2048 x 64]
// ---------------------------------------------------------------------------
__global__ __launch_bounds__(256)
void lora_low_kernel(const float* __restrict__ x,
                     const float* __restrict__ loraA,
                     float* __restrict__ low)
{
    int g = blockIdx.x * 256 + threadIdx.x;     // 0 .. 2048*64-1
    int m = g >> 6;
    int r = g & 63;
    const f4v* xp = (const f4v*)(x + (size_t)m * IN_F);
    const f4v* ap = (const f4v*)(loraA + (size_t)r * IN_F);
    float acc = 0.0f;
    #pragma unroll 8
    for (int k = 0; k < IN_F / 4; ++k) {
        f4v a = xp[k], b = ap[k];
        acc += a[0] * b[0] + a[1] * b[1] + a[2] * b[2] + a[3] * b[3];
    }
    low[g] = SCALING * acc;
}

// ---------------------------------------------------------------------------
// Kernel 2: fused GEMM, bf16x3 emulated-fp32 on v_wmma_f32_16x16x32_bf16
// ---------------------------------------------------------------------------
__global__ __launch_bounds__(256)
void lora_gemm_kernel(const float* __restrict__ x,
                      const float* __restrict__ w,
                      const float* __restrict__ loraB,
                      const float* __restrict__ low,
                      float* __restrict__ out)
{
    __shared__ unsigned int sA[2 * A_PART];   // [hi|lo][BM][PITCH]
    __shared__ unsigned int sB[2 * A_PART];   // [hi|lo][BN][PITCH]

    const int tid   = threadIdx.x;
    const int lane  = tid & 31;
    const int wave  = tid >> 5;        // 0..7
    const int waveM = wave >> 1;       // 0..3 -> 32-row strip
    const int waveN = wave & 1;        // 0..1 -> 64-col strip

    // loader mapping: each thread owns one float4 column, 4 rows (stride 32)
    const int c4   = tid & 7;          // float4 index within a 32-float row
    const int row0 = tid >> 3;         // 0..31

    // m fastest so concurrent blocks share W tiles in L2
    const int m0 = blockIdx.x * BM;
    const int n0 = blockIdx.y * BN;

    // uniform (scalar) bases; thread-varying part is a loop-invariant
    // 32-bit float4 index, row-group delta goes into the 24-bit IOFFSET
    const f4v* aBase = (const f4v*)(x + (size_t)m0 * IN_F);
    const f4v* bBase = (const f4v*)(w + (size_t)n0 * IN_F);
    const f4v* aLora = (const f4v*)(low   + (size_t)m0 * RANK);
    const f4v* bLora = (const f4v*)(loraB + (size_t)n0 * RANK);
    const int offBase = row0 * (IN_F / 4) + c4;   // ld = IN_F
    const int offLora = row0 * (RANK / 4) + c4;   // ld = RANK

    f4v av[4], bv[4];

    auto loadBase = [&](int kt) {
        const f4v* ap = aBase + kt * (BK / 4);
        const f4v* bp = bBase + kt * (BK / 4);
        #pragma unroll
        for (int s = 0; s < 4; ++s) {
            av[s] = ap[offBase + s * 32 * (IN_F / 4)];
            bv[s] = bp[offBase + s * 32 * (IN_F / 4)];
        }
    };
    auto loadLora = [&](int lt) {
        const f4v* ap = aLora + lt * (BK / 4);
        const f4v* bp = bLora + lt * (BK / 4);
        #pragma unroll
        for (int s = 0; s < 4; ++s) {
            av[s] = ap[offLora + s * 32 * (RANK / 4)];
            bv[s] = bp[offLora + s * 32 * (RANK / 4)];
        }
    };

    auto storeStage = [&]() {
        #pragma unroll
        for (int s = 0; s < 4; ++s) {
            int r    = row0 + 32 * s;
            int uidx = r * PITCH + c4 * 2;
            unsigned int h0, l0, h1, l1;
            splitPair(av[s][0], av[s][1], h0, l0);
            splitPair(av[s][2], av[s][3], h1, l1);
            sA[uidx]              = h0;
            sA[uidx + 1]          = h1;
            sA[A_PART + uidx]     = l0;
            sA[A_PART + uidx + 1] = l1;
            splitPair(bv[s][0], bv[s][1], h0, l0);
            splitPair(bv[s][2], bv[s][3], h1, l1);
            sB[uidx]              = h0;
            sB[uidx + 1]          = h1;
            sB[A_PART + uidx]     = l0;
            sB[A_PART + uidx + 1] = l1;
        }
    };

    // A fragment 16x32 bf16: lane r=lane&15 holds row r; k-halves by lane>>4
    auto loadA = [&](int part, int mrow) -> v16bf {
        int r   = lane & 15;
        int off = (lane >> 4) * 4;                       // uints (= 8 bf16)
        const unsigned int* p = sA + part * A_PART + (mrow + r) * PITCH + off;
        Frag f;
        f.q0 = *(const uint4v*)p;                        // K = off..off+7
        f.q1 = *(const uint4v*)(p + 8);                  // K = 16+off..16+off+7
        return f.v;
    };

    // B fragment 32x16 bf16: lane c=lane&15 holds column c; k-half by lane>>4
    auto loadB = [&](int part, int ncol) -> v16bf {
        int c  = lane & 15;
        int kh = lane >> 4;
        const unsigned int* p = sB + part * A_PART + (ncol + c) * PITCH + kh * 4;
        Frag f;
        f.q0 = *(const uint4v*)p;                        // K = kh*8 .. +7
        f.q1 = *(const uint4v*)(p + 8);                  // K = 16+kh*8 .. +7
        return f.v;
    };

    v8f acc[2][4];
    #pragma unroll
    for (int i = 0; i < 2; ++i)
        #pragma unroll
        for (int j = 0; j < 4; ++j)
            acc[i][j] = (v8f)0.0f;

    auto compute = [&]() {
        v16bf ah[2], al[2];
        #pragma unroll
        for (int i = 0; i < 2; ++i) {
            ah[i] = loadA(0, waveM * 32 + i * 16);
            al[i] = loadA(1, waveM * 32 + i * 16);
        }
        #pragma unroll
        for (int j = 0; j < 4; ++j) {
            v16bf bh = loadB(0, waveN * 64 + j * 16);
            v16bf bl = loadB(1, waveN * 64 + j * 16);
            #pragma unroll
            for (int i = 0; i < 2; ++i) {
                // x*w ~= xh*wh + xh*wl + xl*wh   (drop xl*wl, ~2^-16 rel.)
                acc[i][j] = __builtin_amdgcn_wmma_f32_16x16x32_bf16(
                    false, ah[i], false, bh, (short)0, acc[i][j], false, false);
                acc[i][j] = __builtin_amdgcn_wmma_f32_16x16x32_bf16(
                    false, ah[i], false, bl, (short)0, acc[i][j], false, false);
                acc[i][j] = __builtin_amdgcn_wmma_f32_16x16x32_bf16(
                    false, al[i], false, bh, (short)0, acc[i][j], false, false);
            }
        }
    };

    // ---- branch-free main loop: compute stage kt, prefetch stage kt+1 ----
    loadBase(0);
    for (int kt = 0; kt < NBASE - 1; ++kt) {
        __syncthreads();
        storeStage();
        __syncthreads();
        loadBase(kt + 1);
        compute();
    }
    // ---- peeled tail: last base stage, then 2 LoRA stages ----
    __syncthreads();
    storeStage();
    __syncthreads();
    loadLora(0);
    compute();

    __syncthreads();
    storeStage();
    __syncthreads();
    loadLora(1);
    compute();

    __syncthreads();
    storeStage();
    __syncthreads();
    compute();

    // epilogue: C/D layout — VGPR v holds row v (lanes 0-15) / row v+8 (16-31)
    // non-temporal: out is write-once, keep W tiles resident in L2
    float* ob = out + (size_t)m0 * OUT_F + n0;
    const int rBase = waveM * 32 + (lane >> 4) * 8;
    const int cBase = waveN * 64 + (lane & 15);
    #pragma unroll
    for (int i = 0; i < 2; ++i) {
        #pragma unroll
        for (int j = 0; j < 4; ++j) {
            #pragma unroll
            for (int v = 0; v < 8; ++v) {
                int idx = (rBase + i * 16 + v) * OUT_F + cBase + j * 16;
                __builtin_nontemporal_store(acc[i][j][v], ob + idx);
            }
        }
    }
}

// ---------------------------------------------------------------------------
extern "C" void kernel_launch(void* const* d_in, const int* in_sizes, int n_in,
                              void* d_out, int out_size, void* d_ws, size_t ws_size,
                              hipStream_t stream) {
    (void)in_sizes; (void)n_in; (void)out_size; (void)ws_size;
    const float* x     = (const float*)d_in[0];   // [1, 2048, 2048]
    const float* wgt   = (const float*)d_in[1];   // [65536, 2048]
    const float* loraA = (const float*)d_in[2];   // [64, 2048]
    const float* loraB = (const float*)d_in[3];   // [65536, 64]
    float*       out   = (float*)d_out;           // [1, 2048, 65536]
    float*       low   = (float*)d_ws;            // [2048, 64] scratch (512 KB)

    // 1) low = SCALING * (x @ A^T)
    lora_low_kernel<<<(M_TOK * RANK) / 256, 256, 0, stream>>>(x, loraA, low);

    // 2) fused out = x @ W^T + low @ B^T   (K-extension GEMM, m fastest)
    dim3 grid(M_TOK / BM, OUT_F / BN);    // (16, 512)
    lora_gemm_kernel<<<grid, 256, 0, stream>>>(x, wgt, loraB, low, out);
}
